// StandaloneGatedGCNLayer_31344671326721
// MI455X (gfx1250) — matrix operand order repaired and checked by reading, compiled, not run on hardware
//
#include <hip/hip_runtime.h>
#include <hip/hip_bf16.h>

// ---------------------------------------------------------------------------
// GatedGCN layer, MI455X (gfx1250, wave32, WMMA bf16->f32 accumulate).
// Tiles are converted f32->bf16 ONCE at load time into LDS; all WMMA A/B
// fragments are pure b128 loads (no per-fragment VALU conversion).
// ---------------------------------------------------------------------------

typedef __attribute__((ext_vector_type(16))) __bf16 v16bf;
typedef __attribute__((ext_vector_type(8)))  float  v8f;

union BF16x16 { v16bf v; unsigned int u[8]; };
union F32x8   { v8f  v; float f[8]; };

__device__ __forceinline__ unsigned short f2bf(float x) {
    unsigned int a = __float_as_uint(x);
    a = a + 0x7FFFu + ((a >> 16) & 1u);          // round-to-nearest-even
    return (unsigned short)(a >> 16);
}
__device__ __forceinline__ unsigned int pack2bf(float lo, float hi) {
    unsigned int a = __float_as_uint(lo); a = a + 0x7FFFu + ((a >> 16) & 1u);
    unsigned int b = __float_as_uint(hi); b = b + 0x7FFFu + ((b >> 16) & 1u);
    return (a >> 16) | (b & 0xFFFF0000u);
}

// A fragment (16x32 bf16) from a bf16 row in LDS.
// lane l holds M=l; element i -> K = 16*(i>>3) + 8*hi + (i&7)
__device__ __forceinline__ BF16x16 a_frag_bf16(const unsigned short* rowp, int kbase, int hi) {
    const uint4* p0 = (const uint4*)(rowp + kbase + 8 * hi);
    const uint4* p1 = (const uint4*)(rowp + kbase + 16 + 8 * hi);
    uint4 a = p0[0], b = p1[0];
    BF16x16 r;
    r.u[0] = a.x; r.u[1] = a.y; r.u[2] = a.z; r.u[3] = a.w;
    r.u[4] = b.x; r.u[5] = b.y; r.u[6] = b.z; r.u[7] = b.w;
    return r;
}

// B fragment (32x16 bf16): lane l holds N=l; element i -> K = kbase + 16*(i>>3) + 8*hi + (i&7).
// Wt is the pre-transposed weight (N rows x Kdim), bf16, in global (L1/L2 hot).
__device__ __forceinline__ BF16x16 b_frag(const unsigned short* Wt, int nrow, int Kdim,
                                          int kbase, int hi) {
    const uint4* q = (const uint4*)(Wt + (size_t)nrow * Kdim + kbase + 8 * hi);
    uint4 a = q[0], b = q[2];
    BF16x16 r;
    r.u[0] = a.x; r.u[1] = a.y; r.u[2] = a.z; r.u[3] = a.w;
    r.u[4] = b.x; r.u[5] = b.y; r.u[6] = b.z; r.u[7] = b.w;
    return r;
}

__device__ __forceinline__ v8f wmma_bf16(const BF16x16& a, const BF16x16& b, v8f c) {
    return __builtin_amdgcn_wmma_f32_16x16x32_bf16(false, a.v, false, b.v,
                                                   (short)0, c, false, false);
}

// ---------------------------------------------------------------------------
// Kernel: transpose + convert a K x Nn f32 weight into Nn x K bf16 (Wt).
// ---------------------------------------------------------------------------
__global__ void k_wt_bf16(const float* __restrict__ W, unsigned short* __restrict__ Wt,
                          int K, int Nn) {
    int i = blockIdx.x * blockDim.x + threadIdx.x;
    if (i < K * Nn) {
        int n = i / K, k = i - n * K;
        Wt[i] = f2bf(W[(size_t)k * Nn + n]);
    }
}

__global__ void k_zero4(float* __restrict__ p, int n4) {
    int i = blockIdx.x * blockDim.x + threadIdx.x;
    if (i < n4) ((float4*)p)[i] = make_float4(0.f, 0.f, 0.f, 0.f);
}

// ---------------------------------------------------------------------------
// Kernel: 4-way node GEMM.  Out_m = x @ W_m + b_m  for m in {A,B,D,E}.
// Block = 256 threads (8 waves) computes a 128x128 tile; x tile converted to
// bf16 in LDS once; A fragments hoisted and reused across all 4 matrices.
// ---------------------------------------------------------------------------
__global__ void k_node_gemm4(const float* __restrict__ x,
                             const unsigned short* __restrict__ WtA,
                             const unsigned short* __restrict__ WtB,
                             const unsigned short* __restrict__ WtD,
                             const unsigned short* __restrict__ WtE,
                             const float* __restrict__ bA, const float* __restrict__ bB,
                             const float* __restrict__ bD, const float* __restrict__ bE,
                             float* __restrict__ Ax, float* __restrict__ Bx,
                             float* __restrict__ Dx, float* __restrict__ Ex,
                             int nRows) {
    extern __shared__ float lds[];
    unsigned short* tb = (unsigned short*)lds;          // 128x128 bf16 = 32 KB
    const int tid = threadIdx.x;
    const int base = blockIdx.x * 128;

    #pragma unroll
    for (int j = 0; j < 16; ++j) {
        int off = (j * 256 + tid) * 4;
        int rr = off >> 7, cc = off & 127;
        int g = base + rr;
        float4 v = (g < nRows) ? *(const float4*)(x + (size_t)g * 128 + cc)
                               : make_float4(0.f, 0.f, 0.f, 0.f);
        uint2 p; p.x = pack2bf(v.x, v.y); p.y = pack2bf(v.z, v.w);
        *(uint2*)(tb + rr * 128 + cc) = p;
    }
    __syncthreads();

    const int wave = tid >> 5, lane = tid & 31, l = lane & 15, hi = lane >> 4;
    const unsigned short* arow = tb + (wave * 16 + l) * 128;

    BF16x16 af[4];
    #pragma unroll
    for (int s = 0; s < 4; ++s) af[s] = a_frag_bf16(arow, s * 32, hi);

    const unsigned short* Wts[4] = {WtA, WtB, WtD, WtE};
    const float* bias[4] = {bA, bB, bD, bE};
    float* outs[4] = {Ax, Bx, Dx, Ex};

    #pragma unroll
    for (int m = 0; m < 4; ++m) {
        F32x8 acc[8];
        #pragma unroll
        for (int t = 0; t < 8; ++t)
            #pragma unroll
            for (int k = 0; k < 8; ++k) acc[t].f[k] = 0.f;
        #pragma unroll
        for (int s = 0; s < 4; ++s) {
            BF16x16 bf[8];
            #pragma unroll
            for (int t = 0; t < 8; ++t) bf[t] = b_frag(Wts[m], 16 * t + l, 128, s * 32, hi);
            #pragma unroll
            for (int t = 0; t < 8; ++t) acc[t].v = wmma_bf16(af[s], bf[t], acc[t].v);
        }
        #pragma unroll
        for (int t = 0; t < 8; ++t) {
            int n = 16 * t + l;
            float bv = bias[m][n];
            #pragma unroll
            for (int r = 0; r < 8; ++r) {
                int row = base + wave * 16 + r + 8 * hi;
                if (row < nRows) outs[m][(size_t)row * 128 + n] = acc[t].f[r] + bv;
            }
        }
    }
}

// ---------------------------------------------------------------------------
// Kernel: edge phase.  Ce = edge_attr@WC (WMMA), then
//   e_ij = Dx[row] + Ex[col] + Ce + bC
//   msg  = edge_w * sigmoid(e_ij) * Bx[col]  -> atomicAdd into aggr[row]
//   e_final = edge_attr + relu(e_ij)         -> d_out edge section
// ---------------------------------------------------------------------------
__global__ void k_edge(const float* __restrict__ edge_attr,
                       const float* __restrict__ edge_w,
                       const int* __restrict__ edge_idx,
                       const unsigned short* __restrict__ WtC,
                       const float* __restrict__ bC,
                       const float* __restrict__ Bx, const float* __restrict__ Dx,
                       const float* __restrict__ Ex,
                       float* __restrict__ aggr, float* __restrict__ e_out,
                       int E) {
    extern __shared__ float lds[];
    float* tf = lds;                                    // 128x128 f32 (64 KB)
    unsigned short* tb = (unsigned short*)(lds + 128 * 128); // 128x128 bf16 (32 KB)
    const int tid = threadIdx.x;
    const int base = blockIdx.x * 128;

    #pragma unroll
    for (int j = 0; j < 16; ++j) {
        int off = (j * 256 + tid) * 4;
        int rr = off >> 7, cc = off & 127;
        int e = base + rr;
        float4 v = (e < E) ? *(const float4*)(edge_attr + (size_t)e * 128 + cc)
                           : make_float4(0.f, 0.f, 0.f, 0.f);
        *(float4*)(tf + rr * 128 + cc) = v;
        uint2 p; p.x = pack2bf(v.x, v.y); p.y = pack2bf(v.z, v.w);
        *(uint2*)(tb + rr * 128 + cc) = p;
    }
    __syncthreads();

    const int wave = tid >> 5, lane = tid & 31, l = lane & 15, hi = lane >> 4;
    const int m0 = wave * 16;
    const unsigned short* arow = tb + (m0 + l) * 128;

    BF16x16 af[4];
    #pragma unroll
    for (int s = 0; s < 4; ++s) af[s] = a_frag_bf16(arow, s * 32, hi);

    F32x8 acc[8];
    #pragma unroll
    for (int t = 0; t < 8; ++t)
        #pragma unroll
        for (int k = 0; k < 8; ++k) acc[t].f[k] = 0.f;

    #pragma unroll
    for (int s = 0; s < 4; ++s) {
        BF16x16 bf[8];
        #pragma unroll
        for (int t = 0; t < 8; ++t) bf[t] = b_frag(WtC, 16 * t + l, 128, s * 32, hi);
        #pragma unroll
        for (int t = 0; t < 8; ++t) acc[t].v = wmma_bf16(af[s], bf[t], acc[t].v);
    }

    #pragma unroll
    for (int r = 0; r < 8; ++r) {
        int mm = r + 8 * hi;
        int e = base + m0 + mm;
        if (e >= E) continue;
        int vrow = edge_idx[e];
        int vcol = edge_idx[E + e];
        float ew = edge_w[e];
        #pragma unroll
        for (int t = 0; t < 8; ++t) {
            int n = 16 * t + l;
            float eij = acc[t].f[r] + Dx[(size_t)vrow * 128 + n]
                                    + Ex[(size_t)vcol * 128 + n] + bC[n];
            float sig = 1.0f / (1.0f + __expf(-eij));
            float msg = ew * sig * Bx[(size_t)vcol * 128 + n];
            atomicAdd(&aggr[(size_t)vrow * 128 + n], msg);
            e_out[(size_t)e * 128 + n] = tf[(m0 + mm) * 128 + n] + fmaxf(eij, 0.f);
        }
    }
}

// ---------------------------------------------------------------------------
// Kernel: node finalize + fused FFN.
//   x_mid = x + relu(Ax + aggr)            (f32 + bf16 tiles in LDS)
//   h     = relu(x_mid @ W1 + b1)          (bf16 tile in LDS, 128x256)
//   out   = x_mid + h @ W2 + b2            -> d_out node section
// ---------------------------------------------------------------------------
__global__ void k_node_final(const float* __restrict__ x,
                             const float* __restrict__ Ax,
                             const float* __restrict__ aggr,
                             const unsigned short* __restrict__ W1t,  // 256 x 128
                             const float* __restrict__ b1,
                             const unsigned short* __restrict__ W2t,  // 128 x 256
                             const float* __restrict__ b2,
                             float* __restrict__ x_out, int nRows) {
    extern __shared__ float lds[];
    float* xm = lds;                                        // 128x128 f32 (64 KB)
    unsigned short* xb = (unsigned short*)(lds + 128 * 128);// 128x128 bf16 (32 KB)
    unsigned short* h  = xb + 128 * 128;                    // 128x256 bf16 (64 KB)
    const int tid = threadIdx.x;
    const int base = blockIdx.x * 128;

    #pragma unroll
    for (int j = 0; j < 16; ++j) {
        int off = (j * 256 + tid) * 4;
        int rr = off >> 7, cc = off & 127;
        int g = base + rr;
        float4 v;
        if (g < nRows) {
            float4 xv = *(const float4*)(x + (size_t)g * 128 + cc);
            float4 av = *(const float4*)(Ax + (size_t)g * 128 + cc);
            float4 gv = *(const float4*)(aggr + (size_t)g * 128 + cc);
            v.x = xv.x + fmaxf(av.x + gv.x, 0.f);
            v.y = xv.y + fmaxf(av.y + gv.y, 0.f);
            v.z = xv.z + fmaxf(av.z + gv.z, 0.f);
            v.w = xv.w + fmaxf(av.w + gv.w, 0.f);
        } else v = make_float4(0.f, 0.f, 0.f, 0.f);
        *(float4*)(xm + rr * 128 + cc) = v;
        uint2 p; p.x = pack2bf(v.x, v.y); p.y = pack2bf(v.z, v.w);
        *(uint2*)(xb + rr * 128 + cc) = p;
    }
    __syncthreads();

    const int wave = tid >> 5, lane = tid & 31, l = lane & 15, hi = lane >> 4;
    const int m0 = wave * 16;
    const unsigned short* arow = xb + (m0 + l) * 128;

    BF16x16 af[4];
    #pragma unroll
    for (int s = 0; s < 4; ++s) af[s] = a_frag_bf16(arow, s * 32, hi);

    // ---- GEMM1: h = relu(x_mid @ W1 + b1), 256 output cols in two halves ----
    #pragma unroll
    for (int half = 0; half < 2; ++half) {
        F32x8 acc[8];
        #pragma unroll
        for (int t = 0; t < 8; ++t)
            #pragma unroll
            for (int k = 0; k < 8; ++k) acc[t].f[k] = 0.f;
        #pragma unroll
        for (int s = 0; s < 4; ++s) {
            BF16x16 bf[8];
            #pragma unroll
            for (int t = 0; t < 8; ++t)
                bf[t] = b_frag(W1t, half * 128 + 16 * t + l, 128, s * 32, hi);
            #pragma unroll
            for (int t = 0; t < 8; ++t) acc[t].v = wmma_bf16(af[s], bf[t], acc[t].v);
        }
        #pragma unroll
        for (int t = 0; t < 8; ++t) {
            int n = half * 128 + 16 * t + l;
            float bv = b1[n];
            #pragma unroll
            for (int r = 0; r < 8; ++r) {
                int mm = r + 8 * hi;
                h[(m0 + mm) * 256 + n] = f2bf(fmaxf(acc[t].f[r] + bv, 0.f));
            }
        }
    }
    __syncthreads();

    // ---- GEMM2: out = x_mid + h @ W2 + b2  (K = 256) ----
    const unsigned short* hrow = h + (m0 + l) * 256;
    F32x8 acc[8];
    #pragma unroll
    for (int t = 0; t < 8; ++t)
        #pragma unroll
        for (int k = 0; k < 8; ++k) acc[t].f[k] = 0.f;
    #pragma unroll
    for (int s = 0; s < 8; ++s) {
        BF16x16 af2 = a_frag_bf16(hrow, s * 32, hi);
        BF16x16 bf[8];
        #pragma unroll
        for (int t = 0; t < 8; ++t) bf[t] = b_frag(W2t, 16 * t + l, 256, s * 32, hi);
        #pragma unroll
        for (int t = 0; t < 8; ++t) acc[t].v = wmma_bf16(af2, bf[t], acc[t].v);
    }
    #pragma unroll
    for (int t = 0; t < 8; ++t) {
        int n = 16 * t + l;
        float bv = b2[n];
        #pragma unroll
        for (int r = 0; r < 8; ++r) {
            int mm = r + 8 * hi;
            int row = base + m0 + mm;
            if (row < nRows)
                x_out[(size_t)row * 128 + n] = xm[(m0 + mm) * 128 + n] + acc[t].f[r] + bv;
        }
    }
}

// ---------------------------------------------------------------------------
extern "C" void kernel_launch(void* const* d_in, const int* in_sizes, int n_in,
                              void* d_out, int out_size, void* d_ws, size_t ws_size,
                              hipStream_t stream) {
    const float* x         = (const float*)d_in[0];
    const float* edge_attr = (const float*)d_in[1];
    const float* edge_w    = (const float*)d_in[2];
    const float* WA = (const float*)d_in[3];  const float* bA = (const float*)d_in[4];
    const float* WB = (const float*)d_in[5];  const float* bB = (const float*)d_in[6];
    const float* WC = (const float*)d_in[7];  const float* bC = (const float*)d_in[8];
    const float* WD = (const float*)d_in[9];  const float* bD = (const float*)d_in[10];
    const float* WE = (const float*)d_in[11]; const float* bE = (const float*)d_in[12];
    const float* W1 = (const float*)d_in[13]; const float* b1 = (const float*)d_in[14];
    const float* W2 = (const float*)d_in[15]; const float* b2 = (const float*)d_in[16];
    const int*   edge_idx = (const int*)d_in[17];

    const int N = in_sizes[0] / 128;
    const int E = in_sizes[1] / 128;
    const size_t NX = (size_t)N * 128;

    // workspace layout: 5 f32 node tables, then bf16 transposed weights
    float* Ax   = (float*)d_ws;
    float* Bx   = Ax + NX;
    float* Dx   = Bx + NX;
    float* Ex   = Dx + NX;
    float* agg  = Ex + NX;
    unsigned short* WtA = (unsigned short*)(agg + NX);
    unsigned short* WtB = WtA + 128 * 128;
    unsigned short* WtC = WtB + 128 * 128;
    unsigned short* WtD = WtC + 128 * 128;
    unsigned short* WtE = WtD + 128 * 128;
    unsigned short* W1t = WtE + 128 * 128;   // 256 x 128
    unsigned short* W2t = W1t + 256 * 128;   // 128 x 256

    float* x_out = (float*)d_out;
    float* e_out = x_out + NX;

    // 1) weight prep (tiny)
    k_wt_bf16<<<64, 256, 0, stream>>>(WA, WtA, 128, 128);
    k_wt_bf16<<<64, 256, 0, stream>>>(WB, WtB, 128, 128);
    k_wt_bf16<<<64, 256, 0, stream>>>(WC, WtC, 128, 128);
    k_wt_bf16<<<64, 256, 0, stream>>>(WD, WtD, 128, 128);
    k_wt_bf16<<<64, 256, 0, stream>>>(WE, WtE, 128, 128);
    k_wt_bf16<<<128, 256, 0, stream>>>(W1, W1t, 128, 256);
    k_wt_bf16<<<128, 256, 0, stream>>>(W2, W2t, 256, 128);

    // 2) zero the segment-sum accumulator
    int n4 = (int)(NX / 4);
    k_zero4<<<(n4 + 255) / 256, 256, 0, stream>>>(agg, n4);

    // 3) node GEMMs (Ax, Bx, Dx, Ex)
    int nodeTiles = (N + 127) / 128;
    k_node_gemm4<<<nodeTiles, 256, 32 * 1024, stream>>>(
        x, WtA, WtB, WtD, WtE, bA, bB, bD, bE, Ax, Bx, Dx, Ex, N);

    // 4) edge phase (Ce GEMM + gate + scatter-add + e_final)
    int edgeTiles = (E + 127) / 128;
    k_edge<<<edgeTiles, 256, 96 * 1024, stream>>>(
        edge_attr, edge_w, edge_idx, WtC, bC, Bx, Dx, Ex, agg, e_out, E);

    // 5) node finalize + FFN
    k_node_final<<<nodeTiles, 256, 160 * 1024, stream>>>(
        x, Ax, agg, W1t, b1, W2t, b2, x_out, N);
}
